// LlamaAttention_86139864088900
// MI455X (gfx1250) — 16-bit and 8-bit matrix operand order repaired and checked by reading, compiled, not run on hardware
//
#include <hip/hip_runtime.h>
#include <cstdint>
#include <cstddef>

#define B_   2
#define S_   2048
#define H_   2048
#define NH_  16
#define NKV_ 8
#define HD_  128

typedef __attribute__((ext_vector_type(16))) _Float16 v16h;
typedef __attribute__((ext_vector_type(8)))  _Float16 v8h;
typedef __attribute__((ext_vector_type(8)))  float    v8f;
typedef __attribute__((ext_vector_type(8)))  int      v8i;
typedef __attribute__((ext_vector_type(4)))  int      v4i;
typedef __attribute__((ext_vector_type(2)))  int      v2i;

static __device__ __forceinline__ v8f zero8f() {
  v8f r;
#pragma unroll
  for (int i = 0; i < 8; ++i) r[i] = 0.0f;
  return r;
}
static __device__ __forceinline__ v8i zero8i() {
  v8i r;
#pragma unroll
  for (int i = 0; i < 8; ++i) r[i] = 0;
  return r;
}

static __device__ __forceinline__ void async_copy16(unsigned lds_off, const void* gptr) {
  asm volatile("global_load_async_to_lds_b128 %0, %1, off"
               :: "v"(lds_off), "v"(gptr) : "memory");
}
static __device__ __forceinline__ void wait_async0() {
  asm volatile("s_wait_asynccnt 0x0" ::: "memory");
}
static __device__ __forceinline__ void wave_lds_fence() {
  asm volatile("" ::: "memory");
}

__global__ void init_scales_kernel(unsigned* p) {
  if (threadIdx.x < 8) p[threadIdx.x] = 0u;
}

__global__ void amax_kernel(const float* __restrict__ x, size_t n, unsigned* __restrict__ out) {
  __shared__ float red[256];
  float m = 0.0f;
  for (size_t i = (size_t)blockIdx.x * blockDim.x + threadIdx.x; i < n;
       i += (size_t)gridDim.x * blockDim.x)
    m = fmaxf(m, fabsf(x[i]));
  red[threadIdx.x] = m;
  __syncthreads();
  for (int s = 128; s > 0; s >>= 1) {
    if ((int)threadIdx.x < s) red[threadIdx.x] = fmaxf(red[threadIdx.x], red[threadIdx.x + s]);
    __syncthreads();
  }
  if (threadIdx.x == 0) atomicMax(out, __float_as_uint(red[0]));
}

static __device__ __forceinline__ float scale_from_bits(const unsigned* bits) {
  return fmaxf(__uint_as_float(*bits) * (1.0f / 127.0f), 1e-8f);
}

__global__ void quant_kernel(const float* __restrict__ x, size_t n,
                             const unsigned* __restrict__ amax_bits,
                             signed char* __restrict__ q) {
  size_t i = (size_t)blockIdx.x * blockDim.x + threadIdx.x;
  if (i >= n) return;
  float inv = 1.0f / scale_from_bits(amax_bits);
  float r = rintf(x[i] * inv);
  r = fminf(fmaxf(r, -128.0f), 127.0f);
  q[i] = (signed char)(int)r;
}

__global__ __launch_bounds__(32) void gemm_i8_wmma_kernel(
    const signed char* __restrict__ A, const signed char* __restrict__ W,
    float* __restrict__ out, int M, int N, int K,
    const unsigned* __restrict__ sA_bits, const unsigned* __restrict__ sW_bits) {
  int ntn = N >> 6;
  int tile = blockIdx.x;
  int tm = tile / ntn, tn = tile % ntn;
  int lane = threadIdx.x;
  int hi = lane >> 4, lo = lane & 15;

  const signed char* ar0 = A + (size_t)(tm * 32 + lo) * K + hi * 8;
  const signed char* ar1 = ar0 + (size_t)16 * K;
  const signed char* wr  = W + (size_t)(tn * 64 + lo) * K + hi * 8;
  const size_t wstep = (size_t)16 * K;

  v8i acc[2][4];
#pragma unroll
  for (int ms = 0; ms < 2; ++ms)
#pragma unroll
    for (int ns = 0; ns < 4; ++ns) acc[ms][ns] = zero8i();

  for (int k = 0; k < K; k += 64) {
    v8i a[2], bfr[4];
    {
      ((v2i*)&a[0])[0] = *(const v2i*)(ar0 + k);
      ((v2i*)&a[0])[1] = *(const v2i*)(ar0 + k + 16);
      ((v2i*)&a[0])[2] = *(const v2i*)(ar0 + k + 32);
      ((v2i*)&a[0])[3] = *(const v2i*)(ar0 + k + 48);
      ((v2i*)&a[1])[0] = *(const v2i*)(ar1 + k);
      ((v2i*)&a[1])[1] = *(const v2i*)(ar1 + k + 16);
      ((v2i*)&a[1])[2] = *(const v2i*)(ar1 + k + 32);
      ((v2i*)&a[1])[3] = *(const v2i*)(ar1 + k + 48);
    }
#pragma unroll
    for (int ns = 0; ns < 4; ++ns) {
      const signed char* w = wr + (size_t)ns * wstep + k;
      ((v2i*)&bfr[ns])[0] = *(const v2i*)(w);
      ((v2i*)&bfr[ns])[1] = *(const v2i*)(w + 16);
      ((v2i*)&bfr[ns])[2] = *(const v2i*)(w + 32);
      ((v2i*)&bfr[ns])[3] = *(const v2i*)(w + 48);
    }
#pragma unroll
    for (int ms = 0; ms < 2; ++ms)
#pragma unroll
      for (int ns = 0; ns < 4; ++ns)
        acc[ms][ns] = __builtin_amdgcn_wmma_i32_16x16x64_iu8(
            true, a[ms], true, bfr[ns], acc[ms][ns], false, false);
  }

  float s = scale_from_bits(sA_bits) * scale_from_bits(sW_bits);
#pragma unroll
  for (int ms = 0; ms < 2; ++ms)
#pragma unroll
    for (int ns = 0; ns < 4; ++ns)
#pragma unroll
      for (int r = 0; r < 8; ++r) {
        int row = tm * 32 + ms * 16 + r + 8 * hi;
        int col = tn * 64 + ns * 16 + lo;
        out[(size_t)row * N + col] = (float)acc[ms][ns][r] * s;
      }
}

__global__ void rope_pack_kernel(const float* __restrict__ src,
                                 const int* __restrict__ pos,
                                 _Float16* __restrict__ dst, int nheads, float outscale) {
  int idx = blockIdx.x * blockDim.x + threadIdx.x;
  int total = B_ * S_ * nheads * 64;
  if (idx >= total) return;
  int j = idx & 63;
  int t = idx >> 6;
  int h = t % nheads; t /= nheads;
  int s = t % S_;
  int b = t / S_;
  float p = (float)pos[b * S_ + s];
  float inv_freq = __powf(10000.0f, -((float)j) * (1.0f / 64.0f));
  float sn, cs;
  __sincosf(p * inv_freq, &sn, &cs);
  const float* sp = src + (((size_t)b * S_ + s) * nheads + h) * HD_;
  float x0 = sp[j], x1 = sp[j + 64];
  _Float16* dp = dst + (((size_t)b * nheads + h) * S_ + s) * HD_;
  dp[j]      = (_Float16)((x0 * cs - x1 * sn) * outscale);
  dp[j + 64] = (_Float16)((x1 * cs + x0 * sn) * outscale);
}

__global__ void packvt_kernel(const float* __restrict__ v, _Float16* __restrict__ vt) {
  int idx = blockIdx.x * blockDim.x + threadIdx.x;
  if (idx >= B_ * S_ * NKV_ * HD_) return;
  int d = idx & (HD_ - 1);
  int t = idx >> 7;
  int kv = t % NKV_; t /= NKV_;
  int s = t % S_;
  int b = t / S_;
  vt[(((size_t)b * NKV_ + kv) * HD_ + d) * S_ + s] = (_Float16)v[idx];
}

__global__ __launch_bounds__(128) void flash_wmma_kernel(
    const _Float16* __restrict__ Q, const _Float16* __restrict__ Kc,
    const _Float16* __restrict__ Vt, float* __restrict__ attn) {
  __shared__ _Float16 Klds[32 * HD_];
  __shared__ _Float16 Vlds[HD_ * 32];
  __shared__ _Float16 Pb[4][16 * 32];

  int blk = blockIdx.x;
  int nqb = S_ / 64;
  int qb = blk % nqb;
  int t  = blk / nqb;
  int h  = t % NH_;
  int b  = t / NH_;
  int kvh = h >> 1;

  int tid  = threadIdx.x;
  int wave = tid >> 5;
  int lane = tid & 31;
  int hi = lane >> 4, lo = lane & 15;
  int qt = qb * 4 + wave;

  const _Float16* Qp = Q  + (((size_t)b * NH_  + h)   * S_  + qt * 16) * HD_;
  const _Float16* Kp = Kc + ((size_t)b * NKV_ + kvh) * (size_t)S_ * HD_;
  const _Float16* Vp = Vt + ((size_t)b * NKV_ + kvh) * (size_t)HD_ * S_;

  unsigned kbase = (unsigned)(size_t)(void*)&Klds[0];
  unsigned vbase = (unsigned)(size_t)(void*)&Vlds[0];
  _Float16* pw = &Pb[wave][0];

  v16h qa[4];
  {
    const _Float16* qrow = Qp + (size_t)lo * HD_ + hi * 8;
#pragma unroll
    for (int c = 0; c < 4; ++c) {
      ((v8h*)&qa[c])[0] = *(const v8h*)(qrow + c * 32);
      ((v8h*)&qa[c])[1] = *(const v8h*)(qrow + c * 32 + 16);
    }
  }

  v8f o[8];
#pragma unroll
  for (int j = 0; j < 8; ++j) o[j] = zero8f();
  float m[8], l[8];
#pragma unroll
  for (int r = 0; r < 8; ++r) { m[r] = -1e30f; l[r] = 0.0f; }

  int nkb = (qb * 64 + 64) >> 5;
  int qmax = qt * 16 + 15;

  for (int kb = 0; kb < nkb; ++kb) {
    int k0 = kb * 32;

#pragma unroll
    for (int it = 0; it < 4; ++it) {
      int cid = tid + it * 128;
      int krow = cid >> 4, koff = (cid & 15) * 16;
      async_copy16(kbase + (unsigned)(krow * 256 + koff),
                   (const char*)(Kp + (size_t)(k0 + krow) * HD_) + koff);
      int vrow = cid >> 2, voff = (cid & 3) * 16;
      async_copy16(vbase + (unsigned)(vrow * 64 + voff),
                   (const char*)(Vp + (size_t)vrow * S_ + k0) + voff);
    }
    wait_async0();
    __syncthreads();

    bool active = (k0 <= qmax);
    if (active) {
      v8f st[2];
#pragma unroll
      for (int tt = 0; tt < 2; ++tt) {
        const _Float16* krow = &Klds[(size_t)(tt * 16 + lo) * HD_ + hi * 8];
        v8f s = zero8f();
#pragma unroll
        for (int c = 0; c < 4; ++c) {
          v16h kbv;
          ((v8h*)&kbv)[0] = *(const v8h*)(krow + c * 32);
          ((v8h*)&kbv)[1] = *(const v8h*)(krow + c * 32 + 16);
          s = __builtin_amdgcn_wmma_f32_16x16x32_f16(false, qa[c], false, kbv,
                                                     (short)0, s, false, false);
        }
        st[tt] = s;
      }

#pragma unroll
      for (int r = 0; r < 8; ++r) {
        int qi = qt * 16 + r + 8 * hi;
        if (k0 + lo > qi)      st[0][r] = -1e30f;
        if (k0 + 16 + lo > qi) st[1][r] = -1e30f;
      }
#pragma unroll
      for (int r = 0; r < 8; ++r) {
        float mx = fmaxf(st[0][r], st[1][r]);
#pragma unroll
        for (int off = 8; off >= 1; off >>= 1) mx = fmaxf(mx, __shfl_xor(mx, off, 32));
        float mn = fmaxf(m[r], mx);
        float alpha = __expf(m[r] - mn);
        float p0 = __expf(st[0][r] - mn);
        float p1 = __expf(st[1][r] - mn);
        float rs = p0 + p1;
#pragma unroll
        for (int off = 8; off >= 1; off >>= 1) rs += __shfl_xor(rs, off, 32);
        l[r] = alpha * l[r] + rs;
        m[r] = mn;
        st[0][r] = p0;
        st[1][r] = p1;
#pragma unroll
        for (int j = 0; j < 8; ++j) o[j][r] *= alpha;
      }

#pragma unroll
      for (int r = 0; r < 8; ++r) {
        int row = r + 8 * hi;
        pw[row * 32 + lo]      = (_Float16)st[0][r];
        pw[row * 32 + 16 + lo] = (_Float16)st[1][r];
      }
      wave_lds_fence();
      v16h pa;
      {
        const _Float16* pr = &pw[lo * 32 + hi * 8];
        ((v8h*)&pa)[0] = *(const v8h*)(pr);
        ((v8h*)&pa)[1] = *(const v8h*)(pr + 16);
      }

#pragma unroll
      for (int j = 0; j < 8; ++j) {
        const _Float16* vrow = &Vlds[(size_t)(j * 16 + lo) * 32 + hi * 8];
        v16h vb;
        ((v8h*)&vb)[0] = *(const v8h*)(vrow);
        ((v8h*)&vb)[1] = *(const v8h*)(vrow + 16);
        o[j] = __builtin_amdgcn_wmma_f32_16x16x32_f16(false, pa, false, vb,
                                                      (short)0, o[j], false, false);
      }
    }
    __syncthreads();
  }

#pragma unroll
  for (int r = 0; r < 8; ++r) {
    float invl = 1.0f / l[r];
    int row = qt * 16 + r + 8 * hi;
    float* op = attn + ((size_t)b * S_ + row) * (size_t)H_ + h * HD_;
#pragma unroll
    for (int j = 0; j < 8; ++j) op[j * 16 + lo] = o[j][r] * invl;
  }
}

extern "C" void kernel_launch(void* const* d_in, const int* in_sizes, int n_in,
                              void* d_out, int out_size, void* d_ws, size_t ws_size,
                              hipStream_t stream) {
  (void)in_sizes; (void)n_in; (void)out_size; (void)ws_size;

  const float* hidden = (const float*)d_in[0];
  const float* wq     = (const float*)d_in[1];
  const float* wk     = (const float*)d_in[2];
  const float* wv     = (const float*)d_in[3];
  const float* wo     = (const float*)d_in[4];
  const int*   pos    = (const int*)d_in[5];
  float* out          = (float*)d_out;

  char* ws = (char*)d_ws;
  const size_t MB = 1024 * 1024;
  size_t off = 256;
  unsigned* scales = (unsigned*)ws;
  signed char* xq  = (signed char*)(ws + off); off += 8 * MB;
  signed char* wqq = (signed char*)(ws + off); off += 4 * MB;
  signed char* wkq = (signed char*)(ws + off); off += 2 * MB;
  signed char* wvq = (signed char*)(ws + off); off += 2 * MB;
  signed char* woq = (signed char*)(ws + off); off += 4 * MB;
  float* qf        = (float*)(ws + off);       off += 32 * MB;
  float* kf        = (float*)(ws + off);       off += 16 * MB;
  float* vf        = (float*)(ws + off);       off += 16 * MB;
  _Float16* qh     = (_Float16*)(ws + off);    off += 16 * MB;
  _Float16* kh     = (_Float16*)(ws + off);    off += 8 * MB;
  _Float16* vth    = (_Float16*)(ws + off);    off += 8 * MB;
  float* attn      = qf;
  signed char* atq = xq;

  const size_t nX  = (size_t)B_ * S_ * H_;
  const size_t nWq = (size_t)NH_ * HD_ * H_;
  const size_t nWk = (size_t)NKV_ * HD_ * H_;
  const size_t nAt = (size_t)B_ * S_ * NH_ * HD_;
  const int M = B_ * S_;

  auto nb = [](size_t n) { size_t b = (n + 255) / 256; return (int)(b > 16384 ? 16384 : b); };

  init_scales_kernel<<<1, 32, 0, stream>>>(scales);
  amax_kernel<<<nb(nX),  256, 0, stream>>>(hidden, nX,  scales + 0);
  amax_kernel<<<nb(nWq), 256, 0, stream>>>(wq,     nWq, scales + 1);
  amax_kernel<<<nb(nWk), 256, 0, stream>>>(wk,     nWk, scales + 2);
  amax_kernel<<<nb(nWk), 256, 0, stream>>>(wv,     nWk, scales + 3);
  amax_kernel<<<nb(nWq), 256, 0, stream>>>(wo,     nWq, scales + 4);

  quant_kernel<<<(int)((nX  + 255) / 256), 256, 0, stream>>>(hidden, nX,  scales + 0, xq);
  quant_kernel<<<(int)((nWq + 255) / 256), 256, 0, stream>>>(wq,     nWq, scales + 1, wqq);
  quant_kernel<<<(int)((nWk + 255) / 256), 256, 0, stream>>>(wk,     nWk, scales + 2, wkq);
  quant_kernel<<<(int)((nWk + 255) / 256), 256, 0, stream>>>(wv,     nWk, scales + 3, wvq);
  quant_kernel<<<(int)((nWq + 255) / 256), 256, 0, stream>>>(wo,     nWq, scales + 4, woq);

  gemm_i8_wmma_kernel<<<(M / 32) * (2048 / 64), 32, 0, stream>>>(
      xq, wqq, qf, M, 2048, 2048, scales + 0, scales + 1);
  gemm_i8_wmma_kernel<<<(M / 32) * (1024 / 64), 32, 0, stream>>>(
      xq, wkq, kf, M, 1024, 2048, scales + 0, scales + 2);
  gemm_i8_wmma_kernel<<<(M / 32) * (1024 / 64), 32, 0, stream>>>(
      xq, wvq, vf, M, 1024, 2048, scales + 0, scales + 3);

  const float qscale = 0.08838834764831845f;
  rope_pack_kernel<<<(B_ * S_ * NH_ * 64) / 256, 256, 0, stream>>>(qf, pos, qh, NH_, qscale);
  rope_pack_kernel<<<(B_ * S_ * NKV_ * 64) / 256, 256, 0, stream>>>(kf, pos, kh, NKV_, 1.0f);
  packvt_kernel<<<(B_ * S_ * NKV_ * HD_) / 256, 256, 0, stream>>>(vf, vth);

  flash_wmma_kernel<<<B_ * NH_ * (S_ / 64), 128, 0, stream>>>(qh, kh, vth, attn);

  amax_kernel<<<nb(nAt), 256, 0, stream>>>(attn, nAt, scales + 5);
  quant_kernel<<<(int)((nAt + 255) / 256), 256, 0, stream>>>(attn, nAt, scales + 5, atq);
  gemm_i8_wmma_kernel<<<(M / 32) * (2048 / 64), 32, 0, stream>>>(
      atq, woq, out, M, 2048, 2048, scales + 5, scales + 4);
}
